// GPT2Attention_69535520522852
// MI455X (gfx1250) — matrix operand order repaired and checked by reading, not compiled, not run on hardware
//
#include <hip/hip_runtime.h>
#include <hip/hip_bf16.h>

// ---------------------------------------------------------------------------
// GPT-2 attention block on gfx1250 (MI455X).
// bf16 WMMA (fp32 accumulate) + async global->LDS tile pipeline (ASYNCcnt).
// B=2, S=2048, D=1024, H=16, HD=64
// ---------------------------------------------------------------------------

typedef __attribute__((ext_vector_type(16))) __bf16 v16bf;
typedef __attribute__((ext_vector_type(8)))  float  v8f;

union FragU { v16bf v; uint4 q[2]; };

// A-matrix fragment (16x32 bf16): lane 0-15 holds K {0..7, 16..23},
// lane 16-31 holds K {8..15, 24..31} of the 32-wide window `win`.
__device__ __forceinline__ v16bf load_fragA(const unsigned short* win, int hi) {
    FragU u;
    const uint4* p = reinterpret_cast<const uint4*>(win);
    u.q[0] = p[hi];
    u.q[1] = p[2 + hi];
    return u.v;
}

// B-matrix fragment (32x16 bf16): lane 0-15 holds K {0..7, 16..23},
// lane 16-31 holds K {8..15, 24..31} (same order as A).
__device__ __forceinline__ v16bf load_fragB(const unsigned short* win, int hi) {
    FragU u;
    const uint4* p = reinterpret_cast<const uint4*>(win);
    u.q[0] = p[hi];
    u.q[1] = p[2 + hi];
    return u.v;
}

__device__ __forceinline__ v8f wmma_bf16(v16bf a, v16bf b, v8f c) {
    return __builtin_amdgcn_wmma_f32_16x16x32_bf16(
        false, a, false, b, (short)0, c, false, false);
}

// fp32 -> bf16 bits, round-to-nearest-even
__device__ __forceinline__ unsigned short f2bf(float f) {
    unsigned int u = __float_as_uint(f);
    u = (u + 0x7FFFu + ((u >> 16) & 1u)) >> 16;
    return (unsigned short)u;
}

// Raw 32-bit LDS offset of a shared-memory pointer (flat LDS addr low bits).
__device__ __forceinline__ unsigned lds_off(const void* p) {
    return (unsigned)(unsigned long long)(size_t)p;
}

// Async global->LDS copy of 16 bytes per lane (GLOBAL_LOAD_ASYNC_TO_LDS_B128,
// tracked with ASYNCcnt).
__device__ __forceinline__ void async_copy16(unsigned lds, const void* g) {
    asm volatile("global_load_async_to_lds_b128 %0, %1, off"
                 :: "v"(lds), "v"((unsigned long long)(size_t)g)
                 : "memory");
}
__device__ __forceinline__ void wait_async_le4() {
    asm volatile("s_wait_asynccnt 0x4" ::: "memory");
}
__device__ __forceinline__ void wait_async_0() {
    asm volatile("s_wait_asynccnt 0x0" ::: "memory");
}

// ---------------------------------------------------------------------------
// Prep kernels: one-shot fp32 -> bf16 conversion (and weight transpose) so the
// hot loops can use verbatim async global->LDS copies.
// ---------------------------------------------------------------------------
__global__ void __launch_bounds__(256)
cvt_bf16_kernel(const float* __restrict__ src, unsigned short* __restrict__ dst,
                int n) {
    int i = blockIdx.x * 256 + threadIdx.x;
    int stride = gridDim.x * 256;
    for (; i < n; i += stride) dst[i] = f2bf(src[i]);
}

// W[k][n] fp32 -> WT[n][k] bf16, K == 1024 hardcoded (both weights).
__global__ void __launch_bounds__(256)
cvt_transpose_kernel(const float* __restrict__ src, unsigned short* __restrict__ dst,
                     int N, int total) {
    int i = blockIdx.x * 256 + threadIdx.x;
    int stride = gridDim.x * 256;
    for (; i < total; i += stride) {
        int n = i >> 10, k = i & 1023;
        dst[i] = f2bf(src[(size_t)k * N + n]);
    }
}

// ---------------------------------------------------------------------------
// Tiled GEMM: C[M,N] = A[M,K] * WT[N,K]^T + bias, all bf16 operands via WMMA.
// Block tile 128x128, 8 waves; wave w owns m-subtile w and all 8 n-subtiles
// (8 WMMAs per A fragment). Double-buffered LDS fed by async-to-LDS copies.
// MODE 0: fp32 row-major output.
// MODE 1: QKV split epilogue -> Q/K in [BH,S,HD] bf16, V in [BH,HD,S] bf16.
// ---------------------------------------------------------------------------
template <int MODE>
__global__ void __launch_bounds__(256)
gemm_bf16_kernel(const unsigned short* __restrict__ A,
                 const unsigned short* __restrict__ WT,
                 const float* __restrict__ bias,
                 int M, int N, int K,
                 float* __restrict__ Cout,
                 unsigned short* __restrict__ Qo,
                 unsigned short* __restrict__ Ko,
                 unsigned short* __restrict__ Vt) {
    __shared__ __align__(16) unsigned short As[2][128 * 32];  // [m][k] bf16
    __shared__ __align__(16) unsigned short Bs[2][128 * 32];  // [n][k] bf16

    const int t      = threadIdx.x;
    const int lane   = t & 31;
    const int wave   = t >> 5;          // 0..7 == m-subtile
    const int lanelo = lane & 15;
    const int hi     = lane >> 4;
    const int m0     = blockIdx.y * 128;
    const int n0     = blockIdx.x * 128;

    const unsigned asb = lds_off(&As[0][0]);
    const unsigned bsb = lds_off(&Bs[0][0]);

    // Issue one double-buffer stage: 4 async b128 issues per thread
    // (A tile 8KB + B tile 8KB; 128 rows x 4 chunks of 16B each).
    auto issue = [&](int buf, int kk) {
#pragma unroll
        for (int i = 0; i < 2; ++i) {
            int task = t + i * 256;            // 0..511
            int row  = task >> 2, ch = task & 3;
            async_copy16(asb + buf * 8192 + row * 64 + ch * 16,
                         A + (size_t)(m0 + row) * K + kk + ch * 8);
            async_copy16(bsb + buf * 8192 + row * 64 + ch * 16,
                         WT + (size_t)(n0 + row) * K + kk + ch * 8);
        }
    };

    v8f acc[8];
#pragma unroll
    for (int j = 0; j < 8; ++j)
#pragma unroll
        for (int r = 0; r < 8; ++r) acc[j][r] = 0.0f;

    const int nk = K >> 5;
    issue(0, 0);
    for (int i = 0; i < nk; ++i) {
        const int buf = i & 1;
        if (i + 1 < nk) {
            issue(buf ^ 1, (i + 1) << 5);
            wait_async_le4();               // stage i complete (in-order)
        } else {
            wait_async_0();
        }
        __syncthreads();

        const unsigned short* Ab = &As[buf][0];
        const unsigned short* Bb = &Bs[buf][0];
        v16bf a = load_fragA(&Ab[(wave * 16 + lanelo) * 32], hi);
#pragma unroll
        for (int j = 0; j < 8; ++j) {
            v16bf b = load_fragB(&Bb[(j * 16 + lanelo) * 32], hi);
            acc[j] = wmma_bf16(a, b, acc[j]);
        }
        __syncthreads();
    }

    // Epilogue. C layout: vgpr r -> M=r (lanes 0-15) / M=r+8 (lanes 16-31),
    // N = lane&15 inside the 16x16 subtile.
#pragma unroll
    for (int j = 0; j < 8; ++j) {
#pragma unroll
        for (int r = 0; r < 8; ++r) {
            int m = m0 + wave * 16 + r + hi * 8;
            int n = n0 + j * 16 + lanelo;
            float v = acc[j][r] + bias[n];
            if (MODE == 0) {
                Cout[(size_t)m * N + n] = v;
            } else {
                int which = n >> 10;           // 0:Q 1:K 2:V
                int d  = n & 1023;
                int h  = d >> 6, hd = d & 63;
                int b  = m >> 11, s = m & 2047;
                int bh = b * 16 + h;
                if (which == 0)
                    Qo[((size_t)bh * 2048 + s) * 64 + hd] = f2bf(v);
                else if (which == 1)
                    Ko[((size_t)bh * 2048 + s) * 64 + hd] = f2bf(v);
                else // V stored transposed per head: [BH][HD][S]
                    Vt[((size_t)bh * 64 + hd) * 2048 + s] = f2bf(v);
            }
        }
    }
}

// ---------------------------------------------------------------------------
// Causal flash attention. Q/K in [BH,S,64] bf16, V in [BH,64,S] bf16 (already
// transposed). Block: 8 waves / 128 q-rows; key tiles of 64; online softmax
// fp32; QK^T and P*V via bf16 WMMA. Output bf16 [B,S,H*HD] row-major (feeds
// the out-proj GEMM directly).
// ---------------------------------------------------------------------------
__global__ void __launch_bounds__(256)
attn_kernel(const unsigned short* __restrict__ Q,
            const unsigned short* __restrict__ Kp,
            const unsigned short* __restrict__ Vt,
            unsigned short* __restrict__ Out) {
    __shared__ __align__(16) unsigned short Qs[128 * 64];    // [q][d]
    __shared__ __align__(16) unsigned short Ks[64 * 64];     // [key][d]
    __shared__ __align__(16) unsigned short VTs[64 * 64];    // [d][key]
    __shared__ __align__(16) unsigned short Ps[8 * 16 * 64]; // per-wave P

    const int t      = threadIdx.x;
    const int lane   = t & 31;
    const int wave   = t >> 5;
    const int lanelo = lane & 15;
    const int hi     = lane >> 4;
    const int qt     = blockIdx.x;           // 0..15
    const int bh     = blockIdx.y;           // 0..31
    const int q0     = qt * 128;

    const unsigned short* Qg  = Q  + (size_t)bh * 2048 * 64 + (size_t)q0 * 64;
    const unsigned short* Kg0 = Kp + (size_t)bh * 2048 * 64;
    const unsigned short* Vg0 = Vt + (size_t)bh * 64 * 2048;

    const unsigned qsb = lds_off(Qs);
    const unsigned ksb = lds_off(Ks);
    const unsigned vtb = lds_off(VTs);

    // Async-load the whole Q tile (16KB = 1024 x b128 lane-chunks).
#pragma unroll
    for (int i = 0; i < 4; ++i) {
        int task = t + i * 256;
        async_copy16(qsb + task * 16, Qg + (size_t)task * 8);
    }

    v8f   O[4];
    float m_row[8], l_row[8];
#pragma unroll
    for (int j = 0; j < 4; ++j)
#pragma unroll
        for (int r = 0; r < 8; ++r) O[j][r] = 0.0f;
#pragma unroll
    for (int r = 0; r < 8; ++r) { m_row[r] = -3.4e38f; l_row[r] = 0.0f; }

    const int ktiles = (q0 + 128) >> 6;      // causal: tiles up to diagonal
    for (int kt = 0; kt < ktiles; ++kt) {
        // K tile: verbatim 8KB copy.
#pragma unroll
        for (int i = 0; i < 2; ++i) {
            int task = t + i * 256;          // 0..511
            async_copy16(ksb + task * 16,
                         Kg0 + (size_t)kt * 64 * 64 + (size_t)task * 8);
        }
        // V^T tile: 64 rows (d) x 128B from [BH][64][2048].
#pragma unroll
        for (int i = 0; i < 2; ++i) {
            int task = t + i * 256;
            int d = task >> 3, ch = task & 7;
            async_copy16(vtb + d * 128 + ch * 16,
                         Vg0 + (size_t)d * 2048 + kt * 64 + ch * 8);
        }
        wait_async_0();
        __syncthreads();

        // S = Q * K^T (16 x 64 per wave), kdim=64 -> 2 WMMA steps.
        v8f s[4];
#pragma unroll
        for (int ns = 0; ns < 4; ++ns)
#pragma unroll
            for (int r = 0; r < 8; ++r) s[ns][r] = 0.0f;

#pragma unroll
        for (int c = 0; c < 2; ++c) {
            v16bf aq = load_fragA(&Qs[(wave * 16 + lanelo) * 64 + c * 32], hi);
#pragma unroll
            for (int ns = 0; ns < 4; ++ns) {
                v16bf bk = load_fragB(&Ks[(ns * 16 + lanelo) * 64 + c * 32], hi);
                s[ns] = wmma_bf16(aq, bk, s[ns]);
            }
        }

        // Scale 1/sqrt(64), causal mask.
        const int ktbase = kt * 64;
#pragma unroll
        for (int ns = 0; ns < 4; ++ns) {
            int kcol = ktbase + ns * 16 + lanelo;
#pragma unroll
            for (int r = 0; r < 8; ++r) {
                int qrow = q0 + wave * 16 + r + hi * 8;
                float sv = s[ns][r] * 0.125f;
                s[ns][r] = (kcol > qrow) ? -3.0e38f : sv;
            }
        }

        // Online softmax; xor masks 1..8 reduce within each 16-lane half.
#pragma unroll
        for (int r = 0; r < 8; ++r) {
            float mx = -3.4e38f;
#pragma unroll
            for (int ns = 0; ns < 4; ++ns) mx = fmaxf(mx, s[ns][r]);
#pragma unroll
            for (int off = 8; off >= 1; off >>= 1)
                mx = fmaxf(mx, __shfl_xor(mx, off, 32));
            float mnew  = fmaxf(m_row[r], mx);
            float alpha = __expf(m_row[r] - mnew);
            m_row[r] = mnew;
            float sum = 0.0f;
#pragma unroll
            for (int ns = 0; ns < 4; ++ns) {
                float p = __expf(s[ns][r] - mnew);
                s[ns][r] = p;
                sum += p;
                Ps[wave * 1024 + (r + hi * 8) * 64 + ns * 16 + lanelo] = f2bf(p);
            }
#pragma unroll
            for (int off = 8; off >= 1; off >>= 1)
                sum += __shfl_xor(sum, off, 32);
            l_row[r] = l_row[r] * alpha + sum;
#pragma unroll
            for (int ns = 0; ns < 4; ++ns) O[ns][r] *= alpha;
        }
        __syncthreads();

        // O += P * V  (A = P 16x64, B from VTs[d][key]).
#pragma unroll
        for (int c = 0; c < 2; ++c) {
            v16bf ap = load_fragA(&Ps[wave * 1024 + lanelo * 64 + c * 32], hi);
#pragma unroll
            for (int ns = 0; ns < 4; ++ns) {
                v16bf bv = load_fragB(&VTs[(ns * 16 + lanelo) * 64 + c * 32], hi);
                O[ns] = wmma_bf16(ap, bv, O[ns]);
            }
        }
        __syncthreads();
    }

    // Normalize, write bf16 in [B,S,H*HD] row-major ([4096][1024]).
    const int b = bh >> 4, h = bh & 15;
#pragma unroll
    for (int r = 0; r < 8; ++r) {
        float inv = 1.0f / l_row[r];
        int srow = q0 + wave * 16 + r + hi * 8;
#pragma unroll
        for (int ns = 0; ns < 4; ++ns) {
            int d = ns * 16 + lanelo;
            Out[(((size_t)b * 2048 + srow) * 16 + h) * 64 + d] = f2bf(O[ns][r] * inv);
        }
    }
}

// ---------------------------------------------------------------------------
// Launch:
//   prep: Xbf = bf16(X); WqkvT = bf16(Wqkv^T); WpT = bf16(Wp^T)
//   1) QKV GEMM  -> Q,K [BH,S,64] bf16; V [BH,64,S] bf16 (pre-transposed)
//   2) attention -> attn bf16 [4096][1024]
//   3) out-proj GEMM -> d_out fp32
// Workspace layout (bf16 elements unless noted):
//   Xbf 4096*1024 | WqkvT 3072*1024 | WpT 1024*1024
//   Qw 32*2048*64 | Kw same | Vtw same | attn 4096*1024
// ---------------------------------------------------------------------------
extern "C" void kernel_launch(void* const* d_in, const int* in_sizes, int n_in,
                              void* d_out, int out_size, void* d_ws, size_t ws_size,
                              hipStream_t stream) {
    const float* X    = (const float*)d_in[0];   // [2,2048,1024]
    const float* Wqkv = (const float*)d_in[1];   // [1024,3072]
    const float* Bqkv = (const float*)d_in[2];   // [3072]
    const float* Wp   = (const float*)d_in[3];   // [1024,1024]
    const float* Bp   = (const float*)d_in[4];   // [1024]
    float*       out  = (float*)d_out;           // [2,2048,1024] fp32

    unsigned short* p = (unsigned short*)d_ws;
    unsigned short* Xbf   = p;  p += (size_t)4096 * 1024;
    unsigned short* WqkvT = p;  p += (size_t)3072 * 1024;
    unsigned short* WpT   = p;  p += (size_t)1024 * 1024;
    unsigned short* Qw    = p;  p += (size_t)32 * 2048 * 64;
    unsigned short* Kw    = p;  p += (size_t)32 * 2048 * 64;
    unsigned short* Vtw   = p;  p += (size_t)32 * 2048 * 64;
    unsigned short* attn  = p;  // 4096*1024 bf16

    cvt_bf16_kernel<<<2048, 256, 0, stream>>>(X, Xbf, 4096 * 1024);
    cvt_transpose_kernel<<<2048, 256, 0, stream>>>(Wqkv, WqkvT, 3072, 3072 * 1024);
    cvt_transpose_kernel<<<1024, 256, 0, stream>>>(Wp, WpT, 1024, 1024 * 1024);

    // QKV projection: M=4096, N=3072, K=1024; block tile 128x128 -> grid 24x32
    gemm_bf16_kernel<1><<<dim3(24, 32), 256, 0, stream>>>(
        Xbf, WqkvT, Bqkv, 4096, 3072, 1024, nullptr, Qw, Kw, Vtw);

    // Attention: 16 q-tiles x 32 (b,h)
    attn_kernel<<<dim3(16, 32), 256, 0, stream>>>(Qw, Kw, Vtw, attn);

    // Output projection: M=4096, N=1024, K=1024 -> grid 8x32
    gemm_bf16_kernel<0><<<dim3(8, 32), 256, 0, stream>>>(
        attn, WpT, Bp, 4096, 1024, 1024, out, nullptr, nullptr, nullptr);
}
